// CapsuleLayer_69002944578097
// MI455X (gfx1250) — compile-verified, matrix operand order checked by reading, not run on hardware
//
#include <hip/hip_runtime.h>

#define B_   64
#define NI   2048
#define DI   16
#define NO   32
#define DO_  16
#define COLS (NO * DO_)   // 512

typedef __attribute__((ext_vector_type(16))) __bf16 v16bf;
typedef __attribute__((ext_vector_type(2)))  __bf16 v2bf;
typedef __attribute__((ext_vector_type(8)))  float  v8f;

union FragU { unsigned u[8]; v16bf v; };

// Only used in the streaming conversion kernels (bandwidth-bound; VALU free).
__device__ __forceinline__ unsigned pack2_bf16(float a, float b) {
#if __has_builtin(__builtin_amdgcn_cvt_pk_bf16_f32)
  union { v2bf v; unsigned u; } cv;
  cv.v = __builtin_amdgcn_cvt_pk_bf16_f32(a, b);
  return cv.u;
#else
  unsigned ua = __float_as_uint(a), ub = __float_as_uint(b);
  ua = (ua + 0x7FFFu + ((ua >> 16) & 1u)) >> 16;   // RNE f32->bf16
  ub = (ub + 0x7FFFu + ((ub >> 16) & 1u)) >> 16;
  return ua | (ub << 16);
#endif
}

// Frag from pre-converted bf16 rows: one global_load_b128, no VALU conversion.
// K=0..15 real data (lanes 0-15: K0-7, lanes 16-31: K8-15); K=16..31 zero.
__device__ __forceinline__ v16bf load_frag_bf16(const __bf16* __restrict__ base,
                                                size_t row_stride, int row, int koff) {
  uint4 x = *(const uint4*)(base + (size_t)row * row_stride + koff);
  FragU f;
  f.u[0] = x.x; f.u[1] = x.y; f.u[2] = x.z; f.u[3] = x.w;
  f.u[4] = 0u; f.u[5] = 0u; f.u[6] = 0u; f.u[7] = 0u;
  return f.v;
}

// Kernel 0a: inputs f32 -> Xbf bf16 (same [B][NI][DI] layout).
__global__ void __launch_bounds__(256) caps_convx_kernel(const float* __restrict__ inputs,
                                                         unsigned* __restrict__ Xbf) {
  int idx = blockIdx.x * 256 + threadIdx.x;          // over B_*NI*DI/2 = 1,048,576
  if (idx >= B_ * NI * DI / 2) return;
  float2 x = ((const float2*)inputs)[idx];
  Xbf[idx] = pack2_bf16(x.x, x.y);
}

// Kernel 0b: single pass over W (64 MB):
//   Wbf[n][o][d][i]  = bf16(W[...])              (32 MB)
//   Wsumbf[n][o][i]  = bf16(sum_d W[n][o][d][i]) (2 MB)
__global__ void __launch_bounds__(256) caps_wconv_kernel(const float* __restrict__ W,
                                                         unsigned* __restrict__ Wbf,
                                                         unsigned* __restrict__ Wsumbf) {
  int idx = blockIdx.x * 256 + threadIdx.x;          // over NI*NO*DI/2 = 524,288
  if (idx >= NI * NO * DI / 2) return;
  int i0 = (idx & 7) * 2;
  int no = idx >> 3;                                  // n*NO + o
  const float* base = W + (size_t)no * DO_ * DI + i0;
  unsigned* wrow = Wbf + (size_t)no * DO_ * (DI / 2) + (i0 >> 1);
  float s0 = 0.f, s1 = 0.f;
  #pragma unroll
  for (int d = 0; d < DO_; ++d) {
    float a = base[(size_t)d * DI];
    float b = base[(size_t)d * DI + 1];
    s0 += a; s1 += b;
    wrow[(size_t)d * (DI / 2)] = pack2_bf16(a, b);
  }
  Wsumbf[idx] = pack2_bf16(s0, s1);
}

// Kernel 1: fused hat-GEMM + routing + partial s accumulation.
// 512 threads = 16 wave32s; wave w owns o in {2w, 2w+1}.
__global__ void __launch_bounds__(512) caps_main_kernel(
    const __bf16* __restrict__ Xbf, const __bf16* __restrict__ Wbf,
    const __bf16* __restrict__ Wsumbf, float* __restrict__ partial, int G) {
  __shared__ float t_lds[B_ * NO];   // 8 KB
  __shared__ float c_lds[B_ * NO];   // 8 KB

  const int tid   = threadIdx.x;
  const int wave  = tid >> 5;
  const int lane  = tid & 31;
  const int row   = lane & 15;
  const int koff  = (lane < 16) ? 0 : 8;   // K half per lane group
  const int boff  = (lane < 16) ? 0 : 8;   // C/D tile M offset per lane group
  const int chunk = blockIdx.x;
  const int mtw = wave & 3, otw = wave >> 2;   // T-tile for waves 0..7

  v8f sacc[4][2];
  #pragma unroll
  for (int mt = 0; mt < 4; ++mt)
    #pragma unroll
    for (int ol = 0; ol < 2; ++ol)
      sacc[mt][ol] = (v8f){0.f,0.f,0.f,0.f,0.f,0.f,0.f,0.f};

  for (int g = 0; g < G; ++g) {
    const int n = chunk * G + g;

    // A frags: X rows b = 16*mt + row, row stride NI*DI
    v16bf afrag[4];
    #pragma unroll
    for (int mt = 0; mt < 4; ++mt)
      afrag[mt] = load_frag_bf16(Xbf + ((size_t)(16 * mt) * NI + n) * DI,
                                 (size_t)NI * DI, row, koff);

    // hat tiles: wave owns o in {2w, 2w+1}
    v8f hat[4][2];
    #pragma unroll
    for (int ol = 0; ol < 2; ++ol) {
      const int o = 2 * wave + ol;
      v16bf bfrag = load_frag_bf16(Wbf + (size_t)(n * NO + o) * DO_ * DI,
                                   DI, row, koff);
      #pragma unroll
      for (int mt = 0; mt < 4; ++mt) {
        v8f z = (v8f){0.f,0.f,0.f,0.f,0.f,0.f,0.f,0.f};
        hat[mt][ol] = __builtin_amdgcn_wmma_f32_16x16x32_bf16(
            false, afrag[mt], false, bfrag, (short)0, z, false, false);
      }
    }

    // T tiles (waves 0..7): t[b][o] = X @ Wsum^T ; lane = o, VGPR v = b row.
    // Re-load the A frag with mtw folded into the address (L0 hit) instead of
    // dynamically indexing afrag[mtw], which would force cndmask select chains.
    if (wave < 8) {
      v16bf atile = load_frag_bf16(Xbf + ((size_t)(16 * mtw) * NI + n) * DI,
                                   (size_t)NI * DI, row, koff);
      v16bf bsum = load_frag_bf16(Wsumbf + (size_t)(n * NO + 16 * otw) * DI,
                                  DI, row, koff);
      v8f z = (v8f){0.f,0.f,0.f,0.f,0.f,0.f,0.f,0.f};
      v8f tt = __builtin_amdgcn_wmma_f32_16x16x32_bf16(
          false, atile, false, bsum, (short)0, z, false, false);
      const int oo = 16 * otw + row;
      #pragma unroll
      for (int v = 0; v < 8; ++v)
        t_lds[(16 * mtw + boff + v) * NO + oo] = tt[v];
    }

    if (g + 1 < G)
      __builtin_prefetch(Wbf + (size_t)((n + 1) * NO + 2 * wave) * DO_ * DI + lane * 16, 0, 3);

    __syncthreads();

    // Routing (all 512 threads): 8 lanes per b, 4 o's each.
    // b1 = t/32; c1 = softmax(b1); b2 = b1 + c1*t; c = softmax(b2)
    {
      const int b  = tid >> 3;
      const int p4 = (tid & 7) * 4;
      float t4[4], w4[4], e4[4];
      #pragma unroll
      for (int j = 0; j < 4; ++j) t4[j] = t_lds[b * NO + p4 + j];
      float m = -1e30f;
      #pragma unroll
      for (int j = 0; j < 4; ++j) { w4[j] = t4[j] * (1.0f / 32.0f); m = fmaxf(m, w4[j]); }
      m = fmaxf(m, __shfl_xor(m, 1)); m = fmaxf(m, __shfl_xor(m, 2));
      m = fmaxf(m, __shfl_xor(m, 4));
      float s = 0.f;
      #pragma unroll
      for (int j = 0; j < 4; ++j) { e4[j] = __expf(w4[j] - m); s += e4[j]; }
      s += __shfl_xor(s, 1); s += __shfl_xor(s, 2); s += __shfl_xor(s, 4);
      const float inv = 1.0f / s;
      float m2 = -1e30f;
      #pragma unroll
      for (int j = 0; j < 4; ++j) { w4[j] += e4[j] * inv * t4[j]; m2 = fmaxf(m2, w4[j]); }
      m2 = fmaxf(m2, __shfl_xor(m2, 1)); m2 = fmaxf(m2, __shfl_xor(m2, 2));
      m2 = fmaxf(m2, __shfl_xor(m2, 4));
      float s2 = 0.f;
      #pragma unroll
      for (int j = 0; j < 4; ++j) { e4[j] = __expf(w4[j] - m2); s2 += e4[j]; }
      s2 += __shfl_xor(s2, 1); s2 += __shfl_xor(s2, 2); s2 += __shfl_xor(s2, 4);
      const float inv2 = 1.0f / s2;
      #pragma unroll
      for (int j = 0; j < 4; ++j) c_lds[b * NO + p4 + j] = e4[j] * inv2;
    }

    __syncthreads();

    // s += c * hat (c broadcast along d; per VGPR v the b row differs)
    #pragma unroll
    for (int ol = 0; ol < 2; ++ol) {
      const int o = 2 * wave + ol;
      #pragma unroll
      for (int mt = 0; mt < 4; ++mt) {
        #pragma unroll
        for (int v = 0; v < 8; ++v) {
          float cv = c_lds[(16 * mt + boff + v) * NO + o];
          sacc[mt][ol][v] += cv * hat[mt][ol][v];
        }
      }
    }
    __syncthreads();   // protect t_lds/c_lds for next n
  }

  // Write this chunk's partial s: layout [chunk][b][o*16+d]
  float* pbase = partial + (size_t)chunk * B_ * COLS;
  const int d = lane & 15;
  #pragma unroll
  for (int ol = 0; ol < 2; ++ol) {
    const int o = 2 * wave + ol;
    #pragma unroll
    for (int mt = 0; mt < 4; ++mt)
      #pragma unroll
      for (int v = 0; v < 8; ++v)
        pbase[(size_t)(16 * mt + boff + v) * COLS + o * 16 + d] = sacc[mt][ol][v];
  }
}

// Kernel 2: reduce partials over chunks + squash over Do (16-lane groups).
__global__ void __launch_bounds__(256) caps_squash_kernel(
    const float* __restrict__ partial, float* __restrict__ out, int chunks) {
  int idx = blockIdx.x * 256 + threadIdx.x;   // 0..32767 = b*512 + o*16 + d
  float v = 0.f;
  for (int c = 0; c < chunks; ++c) v += partial[(size_t)c * B_ * COLS + idx];
  float s2 = v * v;
  s2 += __shfl_xor(s2, 1);
  s2 += __shfl_xor(s2, 2);
  s2 += __shfl_xor(s2, 4);
  s2 += __shfl_xor(s2, 8);
  float scale = (s2 / (1.0f + s2)) * rsqrtf(s2 + 1e-7f);
  out[idx] = scale * v;
}

extern "C" void kernel_launch(void* const* d_in, const int* in_sizes, int n_in,
                              void* d_out, int out_size, void* d_ws, size_t ws_size,
                              hipStream_t stream) {
  (void)in_sizes; (void)n_in; (void)out_size;
  const float* inputs = (const float*)d_in[0];   // [64,2048,16] f32
  const float* W      = (const float*)d_in[1];   // [1,2048,32,16,16] f32

  const size_t wsumbf_bytes = (size_t)NI * NO * DI * 2;        // 2 MB
  const size_t xbf_bytes    = (size_t)B_ * NI * DI * 2;        // 4 MB
  const size_t wbf_bytes    = (size_t)NI * NO * DO_ * DI * 2;  // 32 MB
  int chunks = 1;
  for (int c = 128; c >= 1; c >>= 1) {
    if ((size_t)c * B_ * COLS * sizeof(float) + wsumbf_bytes + xbf_bytes + wbf_bytes
        <= ws_size) { chunks = c; break; }
  }
  float*  partial = (float*)d_ws;
  char*   p       = (char*)d_ws + (size_t)chunks * B_ * COLS * sizeof(float);
  __bf16* Wsumbf  = (__bf16*)p;
  __bf16* Xbf     = (__bf16*)(p + wsumbf_bytes);
  __bf16* Wbf     = (__bf16*)(p + wsumbf_bytes + xbf_bytes);
  const int G = NI / chunks;

  caps_convx_kernel<<<(B_ * NI * DI / 2 + 255) / 256, 256, 0, stream>>>(
      inputs, (unsigned*)Xbf);
  caps_wconv_kernel<<<(NI * NO * DI / 2 + 255) / 256, 256, 0, stream>>>(
      W, (unsigned*)Wbf, (unsigned*)Wsumbf);
  caps_main_kernel<<<chunks, 512, 0, stream>>>(Xbf, Wbf, Wsumbf, partial, G);
  caps_squash_kernel<<<(B_ * COLS + 255) / 256, 256, 0, stream>>>(
      partial, (float*)d_out, chunks);
}
